// RetNetAngBlock_29326036697180
// MI455X (gfx1250) — compile-verified, matrix operand order checked by reading, not run on hardware
//
#include <hip/hip_runtime.h>
#include <hip/hip_bf16.h>
#include <math.h>

typedef __attribute__((ext_vector_type(16))) _Float16 v16h;
typedef __attribute__((ext_vector_type(8)))  float    v8f;

#define B_      8
#define N_      1024
#define C_      256
#define HEADS_  8
#define D_      32
#define HIDDEN_ 1024

__device__ __forceinline__ int iabs(int a) { return a < 0 ? -a : a; }

// ---------------------------------------------------------------------------
// WMMA fragment helper (CDNA5 ISA 7.12.2 layout, wave32)
// A fragment 16x32 f16, row-major source with leading dim lda:
//   lanes 0-15  : row M = lane,    elems 0..7 = K 0..7,  8..15 = K 16..23
//   lanes 16-31 : row M = lane-16, elems 0..7 = K 8..15, 8..15 = K 24..31
// Same pattern loads a B fragment from a row-major-transposed operand
// Bt[n][k]; per-lane loads stay contiguous (2x b128).
// ---------------------------------------------------------------------------
__device__ __forceinline__ v16h load_frag_a(const _Float16* base, int lda) {
  int lane = threadIdx.x & 31;
  int m  = lane & 15;
  int kg = (lane >> 4) << 3;
  const _Float16* p = base + (size_t)m * lda + kg;
  v16h f;
#pragma unroll
  for (int i = 0; i < 8; ++i) f[i] = p[i];
#pragma unroll
  for (int i = 0; i < 8; ++i) f[i + 8] = p[i + 16];
  return f;
}

__device__ __forceinline__ v8f wmma_f16(v16h a, v16h b, v8f c) {
  return __builtin_amdgcn_wmma_f32_16x16x32_f16(false, a, false, b,
                                                (short)0, c, false, false);
}

// async copy of 16B from global to LDS (ASYNCcnt-tracked); lds_off is the
// byte offset within the wave's LDS allocation.
#define ASYNC_STAGE_B128(ldsoff, gp)                                        \
  asm volatile("global_load_async_to_lds_b128 %0, %1, off"                  \
               :: "v"(ldsoff), "v"(gp) : "memory")
#define WAIT_ASYNC0() asm volatile("s_wait_asynccnt 0x0" ::: "memory")
#define WAIT_DS0()    asm volatile("s_wait_dscnt 0x0" ::: "memory")

// ---------------------------------------------------------------------------
// Weight prep: f32 [K][N] -> f16 transposed [N][K] (coalesced stores)
// ---------------------------------------------------------------------------
__global__ void __launch_bounds__(256)
f32_to_f16_transpose_kernel(const float* __restrict__ src,
                            _Float16* __restrict__ dst, int K, int N) {
  int i = blockIdx.x * 256 + threadIdx.x;
  if (i >= K * N) return;
  int n = i / K, k = i - n * K;
  dst[i] = (_Float16)src[(size_t)k * N + n];
}

// ---------------------------------------------------------------------------
// LayerNorm over C=256, one row per 256-thread block (8 waves, wave32)
// ---------------------------------------------------------------------------
__global__ void __launch_bounds__(256)
ln_kernel(const float* __restrict__ x, const float* __restrict__ g,
          const float* __restrict__ bb, float* __restrict__ o32,
          _Float16* __restrict__ o16) {
  int row = blockIdx.x, tid = threadIdx.x;
  int wid = tid >> 5, lane = tid & 31;
  __shared__ float red[8];
  float v = x[(size_t)row * C_ + tid];

  float s = v;
#pragma unroll
  for (int m = 16; m >= 1; m >>= 1) s += __shfl_xor(s, m, 32);
  if (lane == 0) red[wid] = s;
  __syncthreads();
  float mu = 0.f;
#pragma unroll
  for (int i = 0; i < 8; ++i) mu += red[i];
  mu *= (1.f / C_);
  __syncthreads();

  float d = v - mu;
  float s2 = d * d;
#pragma unroll
  for (int m = 16; m >= 1; m >>= 1) s2 += __shfl_xor(s2, m, 32);
  if (lane == 0) red[wid] = s2;
  __syncthreads();
  float var = 0.f;
#pragma unroll
  for (int i = 0; i < 8; ++i) var += red[i];
  var *= (1.f / C_);

  float y = d * rsqrtf(var + 1e-5f) * g[tid] + bb[tid];
  size_t idx = (size_t)row * C_ + tid;
  if (o32) o32[idx] = y;
  o16[idx] = (_Float16)y;
}

// ---------------------------------------------------------------------------
// WMMA GEMM with async-LDS weight staging.
// Block = 8 waves sharing one 64-column strip, covering 256 rows.
// Per k-tile, the 64x32 f16 weight tile (4KB) is staged into LDS with
// global_load_async_to_lds_b128 (one per thread), double-buffered so the
// next tile's DMA overlaps the current tile's 8 WMMAs per wave.
// ---------------------------------------------------------------------------
template <bool GELU>
__global__ void __launch_bounds__(256)
wmma_gemm_kernel(const _Float16* __restrict__ A, const _Float16* __restrict__ Wt,
                 const float* __restrict__ bias, const float* __restrict__ residual,
                 float* __restrict__ out, int M, int K, int Nc) {
  __shared__ _Float16 lds_w[2][64 * 32];   // 8KB double buffer; LDS offset 0

  int ntRow = Nc >> 6;
  int blkRow = blockIdx.x / ntRow;
  int nt = blockIdx.x - blkRow * ntRow;
  int n0 = nt << 6;
  int wid = threadIdx.x >> 5;
  int row0 = (blkRow << 8) + (wid << 5);   // 32 rows per wave
  int lane = threadIdx.x & 31;
  int nl = lane & 15, mo = (lane >> 4) << 3;

  // staging assignment: thread -> one 16B chunk of the 64x32 tile
  int srow = threadIdx.x >> 2;             // 0..63 (strip row = B column)
  int schunk = (threadIdx.x & 3) << 3;     // 0,8,16,24 (halfs)
  const _Float16* gsrc = Wt + (size_t)(n0 + srow) * K + schunk;
  unsigned lds_elem = (unsigned)(srow * 32 + schunk);
  unsigned lds_off0 = lds_elem * 2u;                 // bytes, buffer 0
  unsigned lds_off1 = lds_off0 + 64u * 32u * 2u;     // bytes, buffer 1

  v8f acc[2][4];
#pragma unroll
  for (int i = 0; i < 2; ++i)
#pragma unroll
    for (int j = 0; j < 4; ++j)
#pragma unroll
      for (int r = 0; r < 8; ++r) acc[i][j][r] = 0.f;

  int ktiles = K >> 5;
  ASYNC_STAGE_B128(lds_off0, gsrc);                  // stage tile 0

  for (int t = 0; t < ktiles; ++t) {
    int k0 = t << 5;
    WAIT_ASYNC0();
    WAIT_DS0();
    __syncthreads();                                 // tile t visible to all
    if (t + 1 < ktiles) {                            // kick tile t+1 DMA
      unsigned dst = (t & 1) ? lds_off0 : lds_off1;
      ASYNC_STAGE_B128(dst, gsrc + ((size_t)(t + 1) << 5));
    }
    if (t + 1 < ktiles) {                            // prefetch next A tile
      __builtin_prefetch(A + (size_t)(row0 + nl) * K + k0 + 32, 0, 0);
      __builtin_prefetch(A + (size_t)(row0 + 16 + nl) * K + k0 + 32, 0, 0);
    }
    const _Float16* lw = &lds_w[t & 1][0];
    v16h a0 = load_frag_a(A + (size_t)row0 * K + k0, K);
    v16h a1 = load_frag_a(A + (size_t)(row0 + 16) * K + k0, K);
#pragma unroll
    for (int j = 0; j < 4; ++j) {
      v16h b = load_frag_a(lw + (size_t)(j << 4) * 32, 32);
      acc[0][j] = wmma_f16(a0, b, acc[0][j]);
      acc[1][j] = wmma_f16(a1, b, acc[1][j]);
    }
  }

#pragma unroll
  for (int i = 0; i < 2; ++i) {
#pragma unroll
    for (int j = 0; j < 4; ++j) {
#pragma unroll
      for (int r = 0; r < 8; ++r) {
        int row = row0 + (i << 4) + mo + r;
        int col = n0 + (j << 4) + nl;
        float v = acc[i][j][r] + bias[col];
        if (GELU) v = 0.5f * v * (1.f + erff(v * 0.70710678118654752f));
        if (residual) v += residual[(size_t)row * Nc + col];
        out[(size_t)row * Nc + col] = v;
      }
    }
  }
}

// ---------------------------------------------------------------------------
// RoPE + scale + layout:
//   q,k -> per-(b,h) row-major [N][32] f16   (A / K^T fragments)
//   v   -> per-(b,h) TRANSPOSED [32][N] f16  (contiguous B fragments for P.V)
// ---------------------------------------------------------------------------
__global__ void __launch_bounds__(256)
rope_prep_kernel(const float* __restrict__ q32, const float* __restrict__ k32,
                 const float* __restrict__ v32, _Float16* __restrict__ q16,
                 _Float16* __restrict__ k16, _Float16* __restrict__ vT) {
  int t = blockIdx.x * 256 + threadIdx.x;   // B*H*N*16 total
  int j = t & 15;
  int n = (t >> 4) & (N_ - 1);
  int bh = t >> 14;
  int h = bh & 7, b = bh >> 3;

  size_t src = ((size_t)b * N_ + n) * C_ + h * D_ + 2 * j;
  float ang = __powf(10000.f, -(float)j * (1.f / 15.f));
  float sn, cs;
  __sincosf((float)n * ang, &sn, &cs);
  const float scale = 0.17677669529663687f;  // 32^-0.5

  float qx = q32[src], qy = q32[src + 1];
  float kx = k32[src], ky = k32[src + 1];
  size_t dst = ((size_t)bh * N_ + n) * D_ + 2 * j;
  q16[dst]     = (_Float16)((qx * cs - qy * sn) * scale);
  q16[dst + 1] = (_Float16)((qy * cs + qx * sn) * scale);
  k16[dst]     = (_Float16)(kx * cs - ky * sn);
  k16[dst + 1] = (_Float16)(ky * cs + kx * sn);
  size_t vdst = ((size_t)bh * D_ + 2 * j) * N_ + n;   // vT[d][n]
  vT[vdst]      = (_Float16)v32[src];
  vT[vdst + N_] = (_Float16)v32[src + 1];
}

// ---------------------------------------------------------------------------
// Flash attention: one wave = 16 query rows of one (b,h); key tiles of 32.
// ---------------------------------------------------------------------------
__global__ void __launch_bounds__(128)
attention_kernel(const _Float16* __restrict__ q16, const _Float16* __restrict__ k16,
                 const _Float16* __restrict__ vT, float* __restrict__ attn_out) {
  __shared__ _Float16 lds_p[4][16 * 32];
  int wid = threadIdx.x >> 5;
  int wave = blockIdx.x * 4 + wid;
  int qt = wave & 63;          // N/16 tiles
  int bh = wave >> 6;
  int h = bh & 7, b = bh >> 3;
  int q0 = qt << 4;

  const _Float16* qm  = q16 + (size_t)bh * N_ * D_;
  const _Float16* km  = k16 + (size_t)bh * N_ * D_;
  const _Float16* vTm = vT  + (size_t)bh * D_ * N_;

  int lane = threadIdx.x & 31;
  int nl = lane & 15, mo = (lane >> 4) << 3;

  v16h aq = load_frag_a(qm + (size_t)q0 * D_, D_);
  float decay = logf(1.f - exp2f(-(1.f + 3.f * (float)h / 8.f)));

  float mi[8], li[8];
  v8f acc0, acc1, zero;
#pragma unroll
  for (int r = 0; r < 8; ++r) {
    mi[r] = -1e30f; li[r] = 0.f;
    acc0[r] = 0.f; acc1[r] = 0.f; zero[r] = 0.f;
  }

  for (int kb = 0; kb < N_; kb += 32) {
    if (kb + 32 < N_) {  // prefetch next tiles
      __builtin_prefetch(km + (size_t)(kb + 32 + nl) * D_, 0, 0);
      __builtin_prefetch(vTm + (size_t)nl * N_ + kb + 32, 0, 0);
      __builtin_prefetch(vTm + (size_t)(16 + nl) * N_ + kb + 32, 0, 0);
    }
    // scores: Q(16x32) x K^T(32x16) twice (keys kb..+15, kb+16..+31)
    v16h bk0 = load_frag_a(km + (size_t)kb * D_, D_);
    v16h bk1 = load_frag_a(km + (size_t)(kb + 16) * D_, D_);
    v8f s0 = wmma_f16(aq, bk0, zero);
    v8f s1 = wmma_f16(aq, bk1, zero);

    int key0 = kb + nl, key1 = kb + 16 + nl;
    int k0h = key0 >> 5, k0w = key0 & 31;
    int k1h = key1 >> 5, k1w = key1 & 31;

#pragma unroll
    for (int r = 0; r < 8; ++r) {
      int qrow = q0 + mo + r;
      int qh = qrow >> 5, qw = qrow & 31;
      s0[r] += decay * (float)(iabs(qh - k0h) + iabs(qw - k0w));
      s1[r] += decay * (float)(iabs(qh - k1h) + iabs(qw - k1w));

      float tm = fmaxf(s0[r], s1[r]);
#pragma unroll
      for (int msk = 8; msk >= 1; msk >>= 1) tm = fmaxf(tm, __shfl_xor(tm, msk, 32));
      float mnew = fmaxf(mi[r], tm);
      float sc = __expf(mi[r] - mnew);
      float e0 = __expf(s0[r] - mnew);
      float e1 = __expf(s1[r] - mnew);
      float ts = e0 + e1;
#pragma unroll
      for (int msk = 8; msk >= 1; msk >>= 1) ts += __shfl_xor(ts, msk, 32);
      li[r] = li[r] * sc + ts;
      mi[r] = mnew;
      acc0[r] *= sc;
      acc1[r] *= sc;
      lds_p[wid][(mo + r) * 32 + nl]      = (_Float16)e0;
      lds_p[wid][(mo + r) * 32 + 16 + nl] = (_Float16)e1;
    }
    __syncthreads();

    // out += P(16x32) x V(32x32); V columns come from transposed vT rows
    v16h ap  = load_frag_a((const _Float16*)lds_p[wid], 32);
    v16h bv0 = load_frag_a(vTm + kb, N_);                       // d 0..15
    v16h bv1 = load_frag_a(vTm + (size_t)16 * N_ + kb, N_);     // d 16..31
    acc0 = wmma_f16(ap, bv0, acc0);
    acc1 = wmma_f16(ap, bv1, acc1);
    __syncthreads();
  }

#pragma unroll
  for (int r = 0; r < 8; ++r) {
    float inv = 1.f / li[r];
    int row = q0 + mo + r;
    size_t o = ((size_t)b * N_ + row) * C_ + h * D_;
    attn_out[o + nl]      = acc0[r] * inv;
    attn_out[o + 16 + nl] = acc1[r] * inv;
  }
}

// ---------------------------------------------------------------------------
// LePE: 5x5 depthwise conv on xn (NHWC, C=256) + add attention out -> f16
// ---------------------------------------------------------------------------
__global__ void __launch_bounds__(256)
lepe_kernel(const float* __restrict__ xn32, const float* __restrict__ w,
            const float* __restrict__ bias, const float* __restrict__ attn32,
            _Float16* __restrict__ comb16) {
  int c = threadIdx.x;
  int bn = blockIdx.x;
  int n = bn & (N_ - 1), b = bn >> 10;
  int y = n >> 5, x = n & 31;
  float s = bias[c];
  const float* wc = w + c * 25;
#pragma unroll
  for (int dy = 0; dy < 5; ++dy) {
    int yy = y + dy - 2;
    if ((unsigned)yy >= 32u) continue;
#pragma unroll
    for (int dx = 0; dx < 5; ++dx) {
      int xx = x + dx - 2;
      if ((unsigned)xx >= 32u) continue;
      s += xn32[(((size_t)b * N_) + yy * 32 + xx) * C_ + c] * wc[dy * 5 + dx];
    }
  }
  size_t idx = (size_t)bn * C_ + c;
  comb16[idx] = (_Float16)(attn32[idx] + s);
}

// ---------------------------------------------------------------------------
// MLP 3x3 depthwise conv residual on h (NHWC, HIDDEN=1024) -> f16
// ---------------------------------------------------------------------------
__global__ void __launch_bounds__(256)
mlpdw_kernel(const float* __restrict__ h32, const float* __restrict__ w,
             const float* __restrict__ bias, _Float16* __restrict__ h2) {
  int bn = blockIdx.x;
  int n = bn & (N_ - 1), b = bn >> 10;
  int y = n >> 5, x = n & 31;
#pragma unroll
  for (int cc = 0; cc < 4; ++cc) {
    int c = threadIdx.x + (cc << 8);
    float hv = h32[(size_t)bn * HIDDEN_ + c];
    float s = bias[c];
    const float* wc = w + c * 9;
#pragma unroll
    for (int dy = 0; dy < 3; ++dy) {
      int yy = y + dy - 1;
      if ((unsigned)yy >= 32u) continue;
#pragma unroll
      for (int dx = 0; dx < 3; ++dx) {
        int xx = x + dx - 1;
        if ((unsigned)xx >= 32u) continue;
        s += h32[(((size_t)b * N_) + yy * 32 + xx) * HIDDEN_ + c] * wc[dy * 3 + dx];
      }
    }
    h2[(size_t)bn * HIDDEN_ + c] = (_Float16)(hv + s);
  }
}

// ---------------------------------------------------------------------------
extern "C" void kernel_launch(void* const* d_in, const int* in_sizes, int n_in,
                              void* d_out, int out_size, void* d_ws, size_t ws_size,
                              hipStream_t stream) {
  (void)in_sizes; (void)n_in; (void)out_size; (void)ws_size;
  const float* x      = (const float*)d_in[0];
  const float* ln1_g  = (const float*)d_in[1];
  const float* ln1_b  = (const float*)d_in[2];
  const float* wq     = (const float*)d_in[3];
  const float* bq     = (const float*)d_in[4];
  const float* wk     = (const float*)d_in[5];
  const float* bk     = (const float*)d_in[6];
  const float* wv     = (const float*)d_in[7];
  const float* bv     = (const float*)d_in[8];
  const float* lepe_w = (const float*)d_in[9];
  const float* lepe_b = (const float*)d_in[10];
  const float* wo     = (const float*)d_in[11];
  const float* bo     = (const float*)d_in[12];
  const float* ln2_g  = (const float*)d_in[13];
  const float* ln2_b  = (const float*)d_in[14];
  const float* fc1_w  = (const float*)d_in[15];
  const float* fc1_b  = (const float*)d_in[16];
  const float* dw_w   = (const float*)d_in[17];
  const float* dw_b   = (const float*)d_in[18];
  const float* fc2_w  = (const float*)d_in[19];
  const float* fc2_b  = (const float*)d_in[20];
  float* out = (float*)d_out;

  char* ws = (char*)d_ws;
  size_t off = 0;
  auto alloc = [&](size_t bytes) {
    size_t o = off;
    off += (bytes + 255) & ~(size_t)255;
    return o;
  };
  const size_t M = (size_t)B_ * N_;  // 8192 rows

  float*    xn32   = (float*)   (ws + alloc(M * C_ * 4));
  _Float16* xn16   = (_Float16*)(ws + alloc(M * C_ * 2));
  _Float16* wq16t  = (_Float16*)(ws + alloc((size_t)C_ * C_ * 2));
  _Float16* wk16t  = (_Float16*)(ws + alloc((size_t)C_ * C_ * 2));
  _Float16* wv16t  = (_Float16*)(ws + alloc((size_t)C_ * C_ * 2));
  _Float16* wo16t  = (_Float16*)(ws + alloc((size_t)C_ * C_ * 2));
  _Float16* fc1t   = (_Float16*)(ws + alloc((size_t)C_ * HIDDEN_ * 2));
  _Float16* fc2t   = (_Float16*)(ws + alloc((size_t)HIDDEN_ * C_ * 2));
  float*    q32    = (float*)   (ws + alloc(M * C_ * 4));
  float*    k32    = (float*)   (ws + alloc(M * C_ * 4));
  float*    v32    = (float*)   (ws + alloc(M * C_ * 4));
  _Float16* q16    = (_Float16*)(ws + alloc(M * C_ * 2));
  _Float16* k16    = (_Float16*)(ws + alloc(M * C_ * 2));
  _Float16* vT     = (_Float16*)(ws + alloc(M * C_ * 2));
  float*    attn32 = (float*)   (ws + alloc(M * C_ * 4));
  _Float16* comb16 = (_Float16*)(ws + alloc(M * C_ * 2));
  float*    x1     = (float*)   (ws + alloc(M * C_ * 4));
  _Float16* xn2_16 = (_Float16*)(ws + alloc(M * C_ * 2));
  float*    h32    = (float*)   (ws + alloc(M * HIDDEN_ * 4));
  _Float16* h2_16  = (_Float16*)(ws + alloc(M * HIDDEN_ * 2));

  auto cvt_t = [&](const float* s, _Float16* d, int K, int N) {
    f32_to_f16_transpose_kernel<<<(K * N + 255) / 256, 256, 0, stream>>>(s, d, K, N);
  };
  auto gemm = [&](const _Float16* A, const _Float16* Wt, const float* bias,
                  const float* res, float* o, int Mm, int K, int Nc, bool gelu) {
    int blocks = (Mm / 256) * (Nc / 64);
    if (gelu)
      wmma_gemm_kernel<true><<<blocks, 256, 0, stream>>>(A, Wt, bias, res, o, Mm, K, Nc);
    else
      wmma_gemm_kernel<false><<<blocks, 256, 0, stream>>>(A, Wt, bias, res, o, Mm, K, Nc);
  };

  // weights -> f16, transposed to [N][K] for contiguous B fragments
  cvt_t(wq, wq16t, C_, C_);
  cvt_t(wk, wk16t, C_, C_);
  cvt_t(wv, wv16t, C_, C_);
  cvt_t(wo, wo16t, C_, C_);
  cvt_t(fc1_w, fc1t, C_, HIDDEN_);
  cvt_t(fc2_w, fc2t, HIDDEN_, C_);

  // LN1
  ln_kernel<<<(int)M, 256, 0, stream>>>(x, ln1_g, ln1_b, xn32, xn16);

  // QKV projections
  gemm(xn16, wq16t, bq, nullptr, q32, (int)M, C_, C_, false);
  gemm(xn16, wk16t, bk, nullptr, k32, (int)M, C_, C_, false);
  gemm(xn16, wv16t, bv, nullptr, v32, (int)M, C_, C_, false);

  // RoPE + scale + head-major layout (v transposed)
  rope_prep_kernel<<<(B_ * HEADS_ * N_ * 16) / 256, 256, 0, stream>>>(
      q32, k32, v32, q16, k16, vT);

  // flash attention: 64 (b,h) x 64 query tiles / 4 waves-per-block
  attention_kernel<<<(B_ * HEADS_ * (N_ / 16)) / 4, 128, 0, stream>>>(
      q16, k16, vT, attn32);

  // LePE conv + combine
  lepe_kernel<<<(int)M, 256, 0, stream>>>(xn32, lepe_w, lepe_b, attn32, comb16);

  // output projection with residual x -> x1
  gemm(comb16, wo16t, bo, x, x1, (int)M, C_, C_, false);

  // LN2
  ln_kernel<<<(int)M, 256, 0, stream>>>(x1, ln2_g, ln2_b, nullptr, xn2_16);

  // FC1 + GELU
  gemm(xn2_16, fc1t, fc1_b, nullptr, h32, (int)M, C_, HIDDEN_, true);

  // 3x3 depthwise residual
  mlpdw_kernel<<<(int)M, 256, 0, stream>>>(h32, dw_w, dw_b, h2_16);

  // FC2 with residual x1 -> final output
  gemm(h2_16, fc2t, fc2_b, x1, out, (int)M, HIDDEN_, C_, false);
}
